// MultiHeadAttention_43413529428366
// MI455X (gfx1250) — compile-verified, matrix operand order checked by reading, not run on hardware
//
#include <hip/hip_runtime.h>
#include <hip/hip_bf16.h>

// ---------------------------------------------------------------------------
// Fused sliding-window MHA for MI455X (gfx1250), wave32 + WMMA f16->f32.
// Pipeline: pack(f16) -> QKV WMMA GEMM (async-LDS double-buffered B panel)
//           -> qk-LN repack -> flash attention (WMMA QK^T/PV, online softmax)
//           -> output WMMA GEMM.
// ---------------------------------------------------------------------------

typedef __attribute__((ext_vector_type(16))) _Float16 v16h;
typedef __attribute__((ext_vector_type(8)))  _Float16 v8h;
typedef __attribute__((ext_vector_type(8)))  float    v8f;

#define B_   2
#define S_   2048
#define E_   1024
#define H_   16
#define D_   64
#define WIN_ 512
#define E3_  (3 * E_)
#define TOK_ (B_ * S_)

// Build a 16x32 f16 WMMA fragment from two contiguous 16-byte chunks.
__device__ __forceinline__ v16h load_frag(const _Float16* p0, const _Float16* p1) {
  v8h lo = *(const v8h*)p0;
  v8h hi = *(const v8h*)p1;
  v16h r;
#pragma unroll
  for (int i = 0; i < 8; ++i) { r[i] = lo[i]; r[8 + i] = hi[i]; }
  return r;
}

__device__ __forceinline__ v8f zero8f() {
  v8f z = {0.f, 0.f, 0.f, 0.f, 0.f, 0.f, 0.f, 0.f};
  return z;
}

// gfx1250 async copy: 16B global -> LDS, tracked by ASYNCcnt.
// (Inline asm: portable across ROCm-7.2 / upstream toolchains, per CDNA5 docs.)
__device__ __forceinline__ void async_copy_b128(void* lds_dst, const void* gsrc) {
  uint32_t loff = (uint32_t)(uintptr_t)lds_dst;  // low 32 bits = LDS byte offset
  uint64_t gaddr = (uint64_t)(uintptr_t)gsrc;
  asm volatile("global_load_async_to_lds_b128 %0, %1, off"
               :: "v"(loff), "v"(gaddr)
               : "memory");
}
__device__ __forceinline__ void wait_async0() {
  asm volatile("s_wait_asynccnt 0" ::: "memory");
}

// ---------------------------------------------------------------------------
// Kernel 0a: f32 -> f16 elementwise pack
// ---------------------------------------------------------------------------
__global__ void mha_pack_f16(const float* __restrict__ src,
                             _Float16* __restrict__ dst, int n) {
  int i = blockIdx.x * blockDim.x + threadIdx.x;
  if (i < n) dst[i] = (_Float16)src[i];
}

// ---------------------------------------------------------------------------
// Kernel 0b: f32 (K x N, row-major) -> f16 transposed (N x K, row-major).
// Makes WMMA B-fragment loads contiguous (column of B == row of B^T).
// ---------------------------------------------------------------------------
__global__ void mha_transpose_f16(const float* __restrict__ src,
                                  _Float16* __restrict__ dst, int K, int N) {
  long i = (long)blockIdx.x * blockDim.x + threadIdx.x;
  long total = (long)K * N;
  if (i < total) {
    int n = (int)(i / K);
    int k = (int)(i % K);
    dst[i] = (_Float16)src[(long)k * N + n];
  }
}

// ---------------------------------------------------------------------------
// WMMA GEMM: C(MxN,f32) = A(MxK,f16) @ Bt^T + bias.  Bt is N x K (pre-transposed).
// Block = 8 waves covering a 128x64 tile. The shared 64-col B panel is staged
// into LDS with gfx1250 async-to-LDS DMA (double-buffered, ASYNCcnt), then
// each wave reads its B fragments via ds_load_b128. One wave = 16x64 tile:
// 1 A-frag (global) + 4 B-frags (LDS) + 4 WMMAs per 32-wide K-step.
// ---------------------------------------------------------------------------
__global__ void __launch_bounds__(256)
mha_gemm_wmma(const _Float16* __restrict__ A,
              const _Float16* __restrict__ Bt,
              const float* __restrict__ bias,
              float* __restrict__ C,
              int M, int N, int K) {
  __shared__ _Float16 ldsB[2][64 * 32];  // two 4 KB B panels (64 cols x 32 k)

  const int tid = threadIdx.x;
  const unsigned lane = tid & 31u;
  const unsigned w    = tid >> 5;
  const int tilesN = N >> 6;
  const int bm = blockIdx.x / tilesN;
  const int bn = blockIdx.x % tilesN;
  const int m0 = bm * 128 + (int)w * 16;
  const int n0 = bn * 64;

  const int koff = (lane < 16) ? 0 : 8;   // K-chunk select per half-wave
  const int mrow = lane & 15;             // A-fragment row owned by this lane
  const int ncol = lane & 15;             // B/C column owned by this lane

  // Panel staging assignment: thread t copies 16B: B^T row (t>>2), k-chunk (t&3)*8
  const int srow = tid >> 2;              // 0..63 (column of B == row of B^T)
  const int sk   = (tid & 3) * 8;         // 0,8,16,24
  const _Float16* bsrc = Bt + (long)(n0 + srow) * K + sk;
  _Float16* sdst[2] = {&ldsB[0][srow * 32 + sk], &ldsB[1][srow * 32 + sk]};

  const _Float16* arow = A + (long)(m0 + mrow) * K;

  v8f acc[4] = {zero8f(), zero8f(), zero8f(), zero8f()};

  // Prologue: DMA panel k=0 into buffer 0.
  async_copy_b128(sdst[0], bsrc);

  for (int k = 0; k < K; k += 32) {
    const int buf = (k >> 5) & 1;
    wait_async0();        // this wave's panel-stage complete
    __syncthreads();      // all waves' stages visible; prior reads of buf^1 done
    if (k + 32 < K)       // DMA next panel into the other buffer (overlaps WMMA)
      async_copy_b128(sdst[buf ^ 1], bsrc + k + 32);

    v16h a = load_frag(arow + k + koff, arow + k + 16 + koff);
    __builtin_prefetch(arow + k + 64, 0, 1);  // global_prefetch_b8 for next A

    const _Float16* panel = &ldsB[buf][0];
#pragma unroll
    for (int nt = 0; nt < 4; ++nt) {
      const _Float16* bbase = panel + (nt * 16 + ncol) * 32;
      v16h b = load_frag(bbase + koff, bbase + 16 + koff);  // 2x ds_load_b128
      acc[nt] = __builtin_amdgcn_wmma_f32_16x16x32_f16(
          false, a, false, b, (short)0, acc[nt], false, false);
    }
  }

  const int rbase = (lane >> 4) * 8;  // C-layout: lanes>=16 hold rows M=r+8
#pragma unroll
  for (int nt = 0; nt < 4; ++nt) {
    int col = n0 + nt * 16 + ncol;
    float bv = bias[col];
#pragma unroll
    for (int r = 0; r < 8; ++r) {
      int row = m0 + rbase + r;
      C[(long)row * N + col] = acc[nt][r] + bv;
    }
  }
}

// ---------------------------------------------------------------------------
// qk-LayerNorm over D=64 + repack. One wave per (b,h,s); 2 elems/lane.
// Q scaled by 1/sqrt(D) here (folds softmax scale into Q).
// Q,K -> [B,H,S,D] f16;  V -> transposed [B,H,D,S] f16 (for PV B-frags).
// ---------------------------------------------------------------------------
__global__ void mha_ln_repack(const float* __restrict__ qkv,
                              const float* __restrict__ qg, const float* __restrict__ qb,
                              const float* __restrict__ kg, const float* __restrict__ kb,
                              _Float16* __restrict__ qh,
                              _Float16* __restrict__ kh,
                              _Float16* __restrict__ vt) {
  const unsigned lane = threadIdx.x & 31u;
  const unsigned wid  = blockIdx.x * (blockDim.x >> 5) + (threadIdx.x >> 5);
  if (wid >= B_ * H_ * S_) return;
  const int s = (int)(wid % S_);
  const int h = (int)((wid / S_) % H_);
  const int b = (int)(wid / (S_ * H_));
  const long tok = (long)b * S_ + s;
  const int d0 = (int)lane * 2, d1 = d0 + 1;

#pragma unroll
  for (int which = 0; which < 2; ++which) {  // 0 = Q, 1 = K
    const float* base = qkv + tok * E3_ + which * E_ + h * D_;
    float x0 = base[d0], x1 = base[d1];
    float sum = x0 + x1;
#pragma unroll
    for (int off = 16; off >= 1; off >>= 1) sum += __shfl_xor(sum, off);
    float mu = sum * (1.f / 64.f);
    float dx0 = x0 - mu, dx1 = x1 - mu;
    float vs = dx0 * dx0 + dx1 * dx1;
#pragma unroll
    for (int off = 16; off >= 1; off >>= 1) vs += __shfl_xor(vs, off);
    float rstd = rsqrtf(vs * (1.f / 64.f) + 1e-5f);
    const float* g = which ? kg : qg;
    const float* bt = which ? kb : qb;
    float scale = which ? 1.f : 0.125f;  // 1/sqrt(64) folded into Q
    float y0 = (dx0 * rstd * g[d0] + bt[d0]) * scale;
    float y1 = (dx1 * rstd * g[d1] + bt[d1]) * scale;
    _Float16* o = (which ? kh : qh) + ((long)(b * H_ + h) * S_ + s) * D_;
    o[d0] = (_Float16)y0;
    o[d1] = (_Float16)y1;
  }
  // V: plain f16, transposed layout [B,H,D,S]
  {
    const float* vbase = qkv + tok * E3_ + 2 * E_ + h * D_;
    _Float16* o = vt + (long)(b * H_ + h) * D_ * S_;
    o[(long)d0 * S_ + s] = (_Float16)vbase[d0];
    o[(long)d1 * S_ + s] = (_Float16)vbase[d1];
  }
}

// ---------------------------------------------------------------------------
// Flash attention (sliding window). One wave per (b,h,16-query block).
// 32 keys/iter: 2 score tiles (2 WMMAs each), mask, online softmax,
// P through per-wave LDS (C-layout -> A-layout), 4 PV WMMAs.
// Output -> attn_out[token][E] f16 (feeds the final WMMA GEMM).
// ---------------------------------------------------------------------------
__global__ void __launch_bounds__(256)
mha_attn_wmma(const _Float16* __restrict__ qh,
              const _Float16* __restrict__ kh,
              const _Float16* __restrict__ vt,
              _Float16* __restrict__ aout) {
  __shared__ _Float16 ldsP[8][16][32];  // per-wave P tile (16q x 32k)
  const unsigned lane = threadIdx.x & 31u;
  const unsigned w    = threadIdx.x >> 5;
  const unsigned wid  = blockIdx.x * (blockDim.x >> 5) + w;
  const int nqb = S_ / 16;
  const int qb = (int)(wid % nqb);
  const int h  = (int)((wid / nqb) % H_);
  const int b  = (int)(wid / (nqb * H_));
  if (b >= B_) return;

  const int q0 = qb * 16;
  const long bh = (long)b * H_ + h;
  const int koff  = (lane < 16) ? 0 : 8;
  const int nloc  = lane & 15;
  const int rbase = (lane >> 4) * 8;

  // Q fragments (A-layout): lane owns query row q0 + nloc, d split 0..31 / 32..63
  const _Float16* qrow = qh + (bh * S_ + q0 + nloc) * D_;
  v16h qa0 = load_frag(qrow + koff,      qrow + 16 + koff);
  v16h qa1 = load_frag(qrow + 32 + koff, qrow + 48 + koff);

  v8f acc[4] = {zero8f(), zero8f(), zero8f(), zero8f()};
  float mrun[8], lrun[8];
#pragma unroll
  for (int r = 0; r < 8; ++r) { mrun[r] = -1e30f; lrun[r] = 0.f; }

  int kstart = q0 - (WIN_ - 1);
  if (kstart < 0) kstart = 0;
  kstart &= ~31;

  for (int kb = kstart; kb <= q0 + 15; kb += 32) {
    // --- QK^T: two 16x16 score tiles over 32 keys ---
    v8f sc[2] = {zero8f(), zero8f()};
#pragma unroll
    for (int t = 0; t < 2; ++t) {
      const _Float16* krow = kh + (bh * S_ + kb + t * 16 + nloc) * D_;
      v16h kf0 = load_frag(krow + koff,      krow + 16 + koff);
      v16h kf1 = load_frag(krow + 32 + koff, krow + 48 + koff);
      sc[t] = __builtin_amdgcn_wmma_f32_16x16x32_f16(
          false, qa0, false, kf0, (short)0, sc[t], false, false);
      sc[t] = __builtin_amdgcn_wmma_f32_16x16x32_f16(
          false, qa1, false, kf1, (short)0, sc[t], false, false);
    }

    // --- mask (causal + window) and row max (half-wave shfl reduce) ---
    float rmax[8];
#pragma unroll
    for (int r = 0; r < 8; ++r) {
      int qi = q0 + rbase + r;
#pragma unroll
      for (int t = 0; t < 2; ++t) {
        int ki = kb + t * 16 + nloc;
        if ((ki > qi) || (qi - ki >= WIN_)) sc[t][r] = -1e30f;
      }
      float v = fmaxf(sc[0][r], sc[1][r]);
#pragma unroll
      for (int off = 1; off < 16; off <<= 1) v = fmaxf(v, __shfl_xor(v, off));
      rmax[r] = v;
    }

    // --- online softmax update ---
#pragma unroll
    for (int r = 0; r < 8; ++r) {
      float mnew  = fmaxf(mrun[r], rmax[r]);
      float alpha = __expf(mrun[r] - mnew);
      mrun[r] = mnew;
      float s = 0.f;
#pragma unroll
      for (int t = 0; t < 2; ++t) {
        float p = (sc[t][r] > -5e29f) ? __expf(sc[t][r] - mnew) : 0.f;
        sc[t][r] = p;
        s += p;
      }
#pragma unroll
      for (int off = 1; off < 16; off <<= 1) s += __shfl_xor(s, off);
      lrun[r] = lrun[r] * alpha + s;
#pragma unroll
      for (int t2 = 0; t2 < 4; ++t2) acc[t2][r] *= alpha;
    }

    // --- P: C-layout -> LDS -> A-layout (in-order LDS within wave) ---
#pragma unroll
    for (int r = 0; r < 8; ++r)
#pragma unroll
      for (int t = 0; t < 2; ++t)
        ldsP[w][rbase + r][t * 16 + nloc] = (_Float16)sc[t][r];

    v16h pa;
    {
      const _Float16* pr = &ldsP[w][nloc][0];
#pragma unroll
      for (int i = 0; i < 8; ++i) {
        pa[i]     = pr[koff + i];
        pa[8 + i] = pr[16 + koff + i];
      }
    }

    // --- PV: V^T rows are contiguous in keys -> clean B-frag loads ---
#pragma unroll
    for (int t2 = 0; t2 < 4; ++t2) {
      const _Float16* vrow = vt + (bh * D_ + t2 * 16 + nloc) * S_ + kb;
      v16h vf = load_frag(vrow + koff, vrow + 16 + koff);
      acc[t2] = __builtin_amdgcn_wmma_f32_16x16x32_f16(
          false, pa, false, vf, (short)0, acc[t2], false, false);
    }
  }

  // --- epilogue: O / l, write f16 token-major for the output GEMM ---
#pragma unroll
  for (int r = 0; r < 8; ++r) {
    float inv = 1.f / lrun[r];
    long tok = (long)b * S_ + q0 + rbase + r;
#pragma unroll
    for (int t2 = 0; t2 < 4; ++t2)
      aout[tok * E_ + h * D_ + t2 * 16 + nloc] = (_Float16)(acc[t2][r] * inv);
  }
}

// ---------------------------------------------------------------------------
// Launch. Workspace layout (bytes):
//   qkv f32   : TOK*3E*4 = 50,331,648
//   qh  f16   :  8,388,608
//   kh  f16   :  8,388,608
//   vt  f16   :  8,388,608   (V transposed [B,H,D,S])
//   xh  f16   :  8,388,608   (x packed; reused as attention output)
//   WqkvT f16 :  6,291,456
//   WoutT f16 :  2,097,152
//   total ~ 88 MB
// ---------------------------------------------------------------------------
extern "C" void kernel_launch(void* const* d_in, const int* in_sizes, int n_in,
                              void* d_out, int out_size, void* d_ws, size_t ws_size,
                              hipStream_t stream) {
  const float* x      = (const float*)d_in[0];
  const float* W_qkv  = (const float*)d_in[1];
  const float* b_qkv  = (const float*)d_in[2];
  const float* q_gam  = (const float*)d_in[3];
  const float* q_bet  = (const float*)d_in[4];
  const float* k_gam  = (const float*)d_in[5];
  const float* k_bet  = (const float*)d_in[6];
  const float* W_out  = (const float*)d_in[7];
  const float* b_out  = (const float*)d_in[8];
  float* out = (float*)d_out;

  char* ws = (char*)d_ws;
  size_t off = 0;
  float*    qkv   = (float*)(ws + off);    off += (size_t)TOK_ * E3_ * 4;
  _Float16* qh    = (_Float16*)(ws + off); off += (size_t)B_ * H_ * S_ * D_ * 2;
  _Float16* kh    = (_Float16*)(ws + off); off += (size_t)B_ * H_ * S_ * D_ * 2;
  _Float16* vt    = (_Float16*)(ws + off); off += (size_t)B_ * H_ * S_ * D_ * 2;
  _Float16* xh    = (_Float16*)(ws + off); off += (size_t)TOK_ * E_ * 2;  // also attn out
  _Float16* WqkvT = (_Float16*)(ws + off); off += (size_t)E3_ * E_ * 2;
  _Float16* WoutT = (_Float16*)(ws + off); off += (size_t)E_ * E_ * 2;
  (void)ws_size; (void)in_sizes; (void)n_in; (void)out_size;

  // 0) pack / transpose weights to f16
  {
    int n = TOK_ * E_;
    mha_pack_f16<<<(n + 255) / 256, 256, 0, stream>>>(x, xh, n);
    long t1 = (long)E_ * E3_;
    mha_transpose_f16<<<(int)((t1 + 255) / 256), 256, 0, stream>>>(W_qkv, WqkvT, E_, E3_);
    long t2 = (long)E_ * E_;
    mha_transpose_f16<<<(int)((t2 + 255) / 256), 256, 0, stream>>>(W_out, WoutT, E_, E_);
  }
  // 1) QKV GEMM: (4096 x 1024) @ (1024 x 3072) + b_qkv -> qkv f32
  {
    int blocks = (TOK_ / 128) * (E3_ / 64);  // 32 * 48 = 1536
    mha_gemm_wmma<<<blocks, 256, 0, stream>>>(xh, WqkvT, b_qkv, qkv, TOK_, E3_, E_);
  }
  // 2) qk-LayerNorm + repack (Q scaled, V transposed)
  {
    int waves = B_ * H_ * S_;  // 65536
    mha_ln_repack<<<waves / 8, 256, 0, stream>>>(qkv, q_gam, q_bet, k_gam, k_bet,
                                                 qh, kh, vt);
  }
  // 3) flash attention -> xh (reused as f16 attention output)
  {
    int waves = B_ * H_ * (S_ / 16);  // 4096
    mha_attn_wmma<<<waves / 8, 256, 0, stream>>>(qh, kh, vt, xh);
  }
  // 4) output GEMM: (4096 x 1024) @ (1024 x 1024) + b_out -> d_out f32
  {
    int blocks = (TOK_ / 128) * (E_ / 64);  // 32 * 16 = 512
    mha_gemm_wmma<<<blocks, 256, 0, stream>>>(xh, WoutT, b_out, out, TOK_, E_, E_);
  }
}
